// TransformerCorrection_15083925144222
// MI455X (gfx1250) — compile-verified, matrix operand order checked by reading, not run on hardware
//
#include <hip/hip_runtime.h>

// ---------------------------------------------------------------------------
// Problem constants (from reference): B=8, D=1024, E=256, H=8, HD=32, L=2
// ---------------------------------------------------------------------------
constexpr int Bc = 8, Dc = 1024, Ec = 256, Hc = 8, HDc = 32, Lc = 2, E2c = 512;
constexpr int NBITSc = 11;
constexpr float NEGc = -1e30f;

typedef __attribute__((ext_vector_type(16))) _Float16 v16h;
typedef __attribute__((ext_vector_type(8)))  float    v8f;

union Frag16 { v16h h; float4 f4[2]; };

// Load a 16-element f16 fragment as two 16B vector loads (global_load_b128 /
// ds_load_b128). Offsets are always 16B aligned by construction.
__device__ __forceinline__ v16h load_frag(const _Float16* p0, const _Float16* p1) {
  Frag16 f;
  f.f4[0] = *reinterpret_cast<const float4*>(p0);
  f.f4[1] = *reinterpret_cast<const float4*>(p1);
  return f.h;
}

__device__ __forceinline__ v8f wmma16(v16h a, v16h b, v8f c) {
  // v_wmma_f32_16x16x32_f16 : D = A(16x32 f16) * B(32x16 f16) + C(16x16 f32)
  return __builtin_amdgcn_wmma_f32_16x16x32_f16(
      /*neg_a=*/false, a, /*neg_b=*/false, b,
      /*c_mod=*/(short)0, c, /*reuse_a=*/false, /*reuse_b=*/false);
}

__device__ __forceinline__ float gelu_exact(float x) {
  return 0.5f * x * (1.0f + erff(x * 0.70710678118654752f));
}

// ---------------------------------------------------------------------------
// K0: per-batch sum of x -> FiLM q_scale[b,d]  (all-but-self mean is analytic)
// ---------------------------------------------------------------------------
__global__ void __launch_bounds__(256) qscale_kernel(
    const float* __restrict__ x, const float* __restrict__ alpha_p,
    const float* __restrict__ a_p, float* __restrict__ qscale) {
  __shared__ float red[256];
  const int b = blockIdx.x, tid = threadIdx.x;
  float s = 0.f;
  for (int j = tid; j < Dc; j += 256) s += x[b * Dc + j];
  red[tid] = s; __syncthreads();
  for (int st = 128; st > 0; st >>= 1) {
    if (tid < st) red[tid] += red[tid + st];
    __syncthreads();
  }
  const float sum = red[0];
  const float ta = tanhf(alpha_p[0]);
  const float fa = a_p[0];
  for (int d = tid; d < Dc; d += 256) {
    float xn = (sum - x[b * Dc + d]) / (float)(Dc - 1);
    float qs = 1.0f + ta * tanhf(fa * xn);
    qscale[b * Dc + d] = fminf(1.3f, fmaxf(0.7f, qs));
  }
}

// ---------------------------------------------------------------------------
// K1: positional stream pos[d,e] (binary codes computed from bit math),
//     h init (= pos for every batch, f32+f16) and kv (f16, for K/V proj).
// ---------------------------------------------------------------------------
__global__ void __launch_bounds__(256) build_pos_kv_kernel(
    const float* __restrict__ x, const float* __restrict__ value_w,
    const float* __restrict__ value_b, const float* __restrict__ fe,
    const float* __restrict__ bw, float* __restrict__ h,
    _Float16* __restrict__ h16, _Float16* __restrict__ kv16) {
  const int idx = blockIdx.x * 256 + threadIdx.x;  // over D*E
  const int d = idx >> 8;
  const int e = idx & 255;
  float pos = fe[idx];
  const int code = d + 1;
#pragma unroll
  for (int j = 0; j < NBITSc; ++j) {
    float bit = (float)((code >> (NBITSc - 1 - j)) & 1) - 0.5f;
    pos += bit * bw[e * NBITSc + j];
  }
#pragma unroll
  for (int b = 0; b < Bc; ++b) {
    size_t o = (size_t)b * Dc * Ec + idx;
    h[o] = pos;
    h16[o] = (_Float16)pos;
    kv16[o] = (_Float16)(x[b * Dc + d] * value_w[e] + value_b[e] + pos);
  }
}

// ---------------------------------------------------------------------------
// K2: f32 -> f16 weight conversion
// ---------------------------------------------------------------------------
__global__ void __launch_bounds__(256) f32_to_f16_kernel(
    const float* __restrict__ src, _Float16* __restrict__ dst, int n) {
  int i = blockIdx.x * 256 + threadIdx.x;
  if (i < n) dst[i] = (_Float16)src[i];
}

// ---------------------------------------------------------------------------
// K3: WMMA GEMM  C[M,N] = A[M,K] @ W[N,K]^T.
//     Each wave owns a 16x64 strip: ONE A fragment reused across 4 B
//     fragments -> 4 back-to-back v_wmma per K-slab (XDL-friendly), 4x fewer
//     address setups and waves than 16x16 tiling.
//     Fused epilogue: +bias[col], *rowscale[row], exact GELU, f32/f16 stores.
// ---------------------------------------------------------------------------
__global__ void __launch_bounds__(256) gemm_wmma_kernel(
    const _Float16* __restrict__ A, const _Float16* __restrict__ W,
    const float* __restrict__ bias, const float* __restrict__ rowscale,
    float* __restrict__ out32, _Float16* __restrict__ out16,
    int M, int N, int K, int fuse_gelu) {
  const int wid = threadIdx.x >> 5;
  const int lane = threadIdx.x & 31;
  const int tile = blockIdx.x * 8 + wid;   // strip id
  const int ntn = N >> 6;                  // strips of 64 cols
  if (tile >= (M >> 4) * ntn) return;      // grids are exact; never diverges
  const int mt = tile / ntn, nt = tile % ntn;
  const int rcol = lane & 15;
  const int kbase = (lane & 16) ? 8 : 0;   // A/B fragment K striping (wave32)
  const int rowAdd = (lane & 16) ? 8 : 0;  // C fragment row striping

  const _Float16* arow = A + (size_t)(mt * 16 + rcol) * K;
  const _Float16* wrow = W + (size_t)(nt * 64 + rcol) * K;
  const size_t wstep = (size_t)16 * K;     // next 16-col group of the strip

  v8f acc0 = {}, acc1 = {}, acc2 = {}, acc3 = {};
  for (int kk = 0; kk < K; kk += 32) {
    v16h af = load_frag(arow + kk + kbase, arow + kk + kbase + 16);
    v16h b0 = load_frag(wrow + kk + kbase, wrow + kk + kbase + 16);
    v16h b1 = load_frag(wrow + wstep + kk + kbase, wrow + wstep + kk + kbase + 16);
    v16h b2 = load_frag(wrow + 2 * wstep + kk + kbase, wrow + 2 * wstep + kk + kbase + 16);
    v16h b3 = load_frag(wrow + 3 * wstep + kk + kbase, wrow + 3 * wstep + kk + kbase + 16);
    if (kk + 32 < K) {  // global_prefetch_b8 next K-slab
      __builtin_prefetch(arow + kk + 32, 0, 1);
      __builtin_prefetch(wrow + kk + 32, 0, 1);
    }
    acc0 = wmma16(af, b0, acc0);
    acc1 = wmma16(af, b1, acc1);
    acc2 = wmma16(af, b2, acc2);
    acc3 = wmma16(af, b3, acc3);
  }

#pragma unroll
  for (int j = 0; j < 4; ++j) {
    const v8f& acc = (j == 0) ? acc0 : (j == 1) ? acc1 : (j == 2) ? acc2 : acc3;
#pragma unroll
    for (int i = 0; i < 8; ++i) {
      const int r = mt * 16 + i + rowAdd;
      const int c = nt * 64 + j * 16 + rcol;
      float v = acc[i];
      if (bias) v += bias[c];
      if (rowscale) v *= rowscale[r];
      if (fuse_gelu) v = gelu_exact(v);
      if (out32) out32[(size_t)r * N + c] = v;
      if (out16) out16[(size_t)r * N + c] = (_Float16)v;
    }
  }
}

// ---------------------------------------------------------------------------
// K4: V -> V^T  ( [B,D,E] -> [B,H,HD,D] ) so attention B-fragments of V are
//     contiguous 16B loads.
// ---------------------------------------------------------------------------
__global__ void __launch_bounds__(256) transpose_v_kernel(
    const _Float16* __restrict__ vp, _Float16* __restrict__ vt) {
  const int idx = blockIdx.x * 256 + threadIdx.x;  // over B*D*E
  const int e = idx & 255;
  const int d = (idx >> 8) & 1023;
  const int b = idx >> 18;
  const int hh = e >> 5, hd = e & 31;
  vt[((size_t)((b * Hc + hh) * HDc) + hd) * Dc + d] = vp[idx];
}

// ---------------------------------------------------------------------------
// K5: flash attention. One wave per (b, h, 16-query tile). 32 keys / iter:
//     2 score WMMAs, analytic ALiBi+mask bias, online softmax (16-wide
//     shfl_xor matches C-layout row striping), P via per-wave LDS tile,
//     2 P@V WMMAs. Never materializes the [B,H,D,D] scores.
// ---------------------------------------------------------------------------
__global__ void __launch_bounds__(256) flash_attn_kernel(
    const _Float16* __restrict__ q16, const _Float16* __restrict__ k16,
    const _Float16* __restrict__ vt16, float* __restrict__ attn,
    const float* __restrict__ alibi_s) {
  __shared__ __align__(16) _Float16 psm[8][16 * 32];  // 1KB per wave
  const int wid = threadIdx.x >> 5;
  const int lane = threadIdx.x & 31;
  const int gid = blockIdx.x * 8 + wid;  // B*H*(D/16) = 4096 tiles
  const int b = gid >> 9;
  const int hh = (gid >> 6) & 7;
  const int q0 = (gid & 63) << 4;
  const int rcol = lane & 15;
  const int kbase = (lane & 16) ? 8 : 0;
  const int rowAdd = (lane & 16) ? 8 : 0;
  const float strength = fmaxf(alibi_s[0], 0.0f);
  const float slope = exp2f(-(float)hh / 8.0f);
  const float invs = 0.17677669529663687f;  // 1/sqrt(HD)

  const _Float16* qb = q16 + ((size_t)(b * Dc + q0 + rcol) * Ec + hh * HDc);
  const v16h aq = load_frag(qb + kbase, qb + kbase + 16);  // Q loaded once

  float rowmax[8], rowsum[8];
  v8f o0 = {}, o1 = {};
#pragma unroll
  for (int i = 0; i < 8; ++i) { rowmax[i] = -3.0e38f; rowsum[i] = 0.0f; }

  for (int c = 0; c < Dc / 32; ++c) {
    const int kc = c * 32;
    const _Float16* kb0 = k16 + ((size_t)(b * Dc + kc + rcol) * Ec + hh * HDc);
    const _Float16* kb1 = kb0 + 16 * Ec;
    v16h bk0 = load_frag(kb0 + kbase, kb0 + kbase + 16);
    v16h bk1 = load_frag(kb1 + kbase, kb1 + kbase + 16);
    v8f s0 = {}, s1 = {};
    s0 = wmma16(aq, bk0, s0);
    s1 = wmma16(aq, bk1, s1);

#pragma unroll
    for (int i = 0; i < 8; ++i) {
      const int qq = q0 + i + rowAdd;
      const int key0 = kc + rcol;
      const int key1 = key0 + 16;
      float x0 = s0[i] * invs + ((qq == key0) ? NEGc : 0.f)
               - strength * slope * fabsf((float)(qq - key0));
      float x1 = s1[i] * invs + ((qq == key1) ? NEGc : 0.f)
               - strength * slope * fabsf((float)(qq - key1));
      float cm = fmaxf(x0, x1);
      for (int off = 8; off >= 1; off >>= 1) cm = fmaxf(cm, __shfl_xor(cm, off, 16));
      float nm = fmaxf(rowmax[i], cm);
      float corr = __expf(rowmax[i] - nm);
      rowmax[i] = nm;
      rowsum[i] *= corr;
      o0[i] *= corr;
      o1[i] *= corr;
      float p0 = __expf(x0 - nm);
      float p1 = __expf(x1 - nm);
      float ps = p0 + p1;
      for (int off = 8; off >= 1; off >>= 1) ps += __shfl_xor(ps, off, 16);
      rowsum[i] += ps;
      const int rowm = i + rowAdd;
      psm[wid][rowm * 32 + rcol] = (_Float16)p0;       // C-layout -> A-layout
      psm[wid][rowm * 32 + 16 + rcol] = (_Float16)p1;  // via LDS staging
    }
    asm volatile("" ::: "memory");  // order DS stores before DS loads
    const _Float16* pb = &psm[wid][rcol * 32];
    v16h ap = load_frag(pb + kbase, pb + kbase + 16);  // ds_load_b128 x2
    const _Float16* vb0 = vt16 + (((size_t)((b * Hc + hh) * HDc) + rcol) * Dc + kc);
    const _Float16* vb1 = vb0 + 16 * Dc;
    v16h bv0 = load_frag(vb0 + kbase, vb0 + kbase + 16);
    v16h bv1 = load_frag(vb1 + kbase, vb1 + kbase + 16);
    o0 = wmma16(ap, bv0, o0);
    o1 = wmma16(ap, bv1, o1);
    asm volatile("" ::: "memory");  // next-iter stores can't hoist over loads
  }

#pragma unroll
  for (int i = 0; i < 8; ++i) {
    const int qq = q0 + i + rowAdd;
    const float inv = 1.0f / rowsum[i];
    float* ob = attn + ((size_t)(b * Dc + qq) * Ec + hh * HDc);
    ob[rcol] = o0[i] * inv;
    ob[16 + rcol] = o1[i] * inv;
  }
}

// ---------------------------------------------------------------------------
// K6: h += gamma * resid (optional); LayerNorm -> y16. One row per block.
// ---------------------------------------------------------------------------
__global__ void __launch_bounds__(256) resid_ln_kernel(
    float* __restrict__ h, const float* __restrict__ resid,
    const float* __restrict__ gamma, const float* __restrict__ lnw,
    const float* __restrict__ lnb, _Float16* __restrict__ y16) {
  __shared__ float red[256];
  const int row = blockIdx.x, tid = threadIdx.x;
  const size_t o = (size_t)row * Ec + tid;
  float v = h[o];
  if (resid) { v += gamma[tid] * resid[o]; h[o] = v; }
  red[tid] = v; __syncthreads();
  for (int st = 128; st > 0; st >>= 1) {
    if (tid < st) red[tid] += red[tid + st];
    __syncthreads();
  }
  const float mean = red[0] / (float)Ec;
  __syncthreads();
  const float dv = v - mean;
  red[tid] = dv * dv; __syncthreads();
  for (int st = 128; st > 0; st >>= 1) {
    if (tid < st) red[tid] += red[tid + st];
    __syncthreads();
  }
  const float var = red[0] / (float)Ec;
  y16[o] = (_Float16)(dv * rsqrtf(var + 1e-5f) * lnw[tid] + lnb[tid]);
}

// ---------------------------------------------------------------------------
// K7: h += gamma_ffn * ffn_out; refresh f16 mirror of h
// ---------------------------------------------------------------------------
__global__ void __launch_bounds__(256) resid_h16_kernel(
    float* __restrict__ h, const float* __restrict__ f,
    const float* __restrict__ gamma, _Float16* __restrict__ h16) {
  const int idx = blockIdx.x * 256 + threadIdx.x;
  const int e = idx & 255;
  float v = h[idx] + gamma[e] * f[idx];
  h[idx] = v;
  h16[idx] = (_Float16)v;
}

// ---------------------------------------------------------------------------
// K8: out[row] = g[row,:] . corr_w2 + corr_b2   (one wave per row, shfl tree)
// ---------------------------------------------------------------------------
__global__ void __launch_bounds__(256) final_dot_kernel(
    const float* __restrict__ g, const float* __restrict__ w2,
    const float* __restrict__ b2, float* __restrict__ out) {
  const int wid = threadIdx.x >> 5, lane = threadIdx.x & 31;
  const int row = blockIdx.x * 8 + wid;
  float acc = 0.f;
  for (int j = lane; j < Ec; j += 32) acc += g[(size_t)row * Ec + j] * w2[j];
  for (int off = 16; off >= 1; off >>= 1) acc += __shfl_xor(acc, off, 32);
  if (lane == 0) out[row] = acc + b2[0];
}

// ---------------------------------------------------------------------------
extern "C" void kernel_launch(void* const* d_in, const int* in_sizes, int n_in,
                              void* d_out, int out_size, void* d_ws, size_t ws_size,
                              hipStream_t stream) {
  const float* x          = (const float*)d_in[0];
  const float* value_w    = (const float*)d_in[1];
  const float* value_b    = (const float*)d_in[2];
  const float* fe         = (const float*)d_in[3];
  const float* bw         = (const float*)d_in[4];
  const float* qw         = (const float*)d_in[5];
  const float* kw         = (const float*)d_in[6];
  const float* vw         = (const float*)d_in[7];
  const float* film_alpha = (const float*)d_in[8];
  const float* film_a     = (const float*)d_in[9];
  const float* ln1_w      = (const float*)d_in[10];
  const float* ln1_b      = (const float*)d_in[11];
  const float* ffn_w1     = (const float*)d_in[12];
  const float* ffn_b1     = (const float*)d_in[13];
  const float* ffn_w2     = (const float*)d_in[14];
  const float* ffn_b2     = (const float*)d_in[15];
  const float* gamma_attn = (const float*)d_in[16];
  const float* gamma_ffn  = (const float*)d_in[17];
  const float* corr_ln_w  = (const float*)d_in[18];
  const float* corr_ln_b  = (const float*)d_in[19];
  const float* corr_w1    = (const float*)d_in[20];
  const float* corr_b1    = (const float*)d_in[21];
  const float* corr_w2    = (const float*)d_in[22];
  const float* corr_b2    = (const float*)d_in[23];
  const float* alibi_s    = (const float*)d_in[24];
  float* out = (float*)d_out;
  (void)in_sizes; (void)n_in; (void)out_size; (void)ws_size;

  // Workspace carve (~49 MB total, fits comfortably in L2)
  char* ws = (char*)d_ws;
  size_t cur = 0;
  auto carve = [&](size_t bytes) -> char* {
    char* p = ws + cur;
    cur += (bytes + 255) & ~(size_t)255;
    return p;
  };
  constexpr size_t BDE = (size_t)Bc * Dc * Ec;
  float*    h     = (float*)carve(BDE * 4);
  _Float16* h16   = (_Float16*)carve(BDE * 2);
  _Float16* kv16  = (_Float16*)carve(BDE * 2);
  _Float16* k16   = (_Float16*)carve(BDE * 2);
  _Float16* vt16  = (_Float16*)carve(BDE * 2);
  _Float16* s16a  = (_Float16*)carve(BDE * 2);   // V-proj, then per-layer Q
  _Float16* y16   = (_Float16*)carve(BDE * 2);
  _Float16* z16   = (_Float16*)carve((size_t)Bc * Dc * E2c * 2);
  float*    f32b  = (float*)carve(BDE * 4);      // attn / ffn-out / head
  _Float16* qw16  = (_Float16*)carve((size_t)Ec * Ec * 2);
  _Float16* kw16  = (_Float16*)carve((size_t)Ec * Ec * 2);
  _Float16* vw16  = (_Float16*)carve((size_t)Ec * Ec * 2);
  _Float16* cw116 = (_Float16*)carve((size_t)Ec * Ec * 2);
  _Float16* fw116 = (_Float16*)carve((size_t)E2c * Ec * 2);
  _Float16* fw216 = (_Float16*)carve((size_t)Ec * E2c * 2);
  float*    qsc   = (float*)carve((size_t)Bc * Dc * 4);

  const int M = Bc * Dc;  // 8192 rows

  qscale_kernel<<<Bc, 256, 0, stream>>>(x, film_alpha, film_a, qsc);
  build_pos_kv_kernel<<<(Dc * Ec) / 256, 256, 0, stream>>>(
      x, value_w, value_b, fe, bw, h, h16, kv16);
  f32_to_f16_kernel<<<(Ec * Ec) / 256, 256, 0, stream>>>(qw, qw16, Ec * Ec);
  f32_to_f16_kernel<<<(Ec * Ec) / 256, 256, 0, stream>>>(kw, kw16, Ec * Ec);
  f32_to_f16_kernel<<<(Ec * Ec) / 256, 256, 0, stream>>>(vw, vw16, Ec * Ec);
  f32_to_f16_kernel<<<(Ec * Ec) / 256, 256, 0, stream>>>(corr_w1, cw116, Ec * Ec);
  f32_to_f16_kernel<<<(E2c * Ec) / 256, 256, 0, stream>>>(ffn_w1, fw116, E2c * Ec);
  f32_to_f16_kernel<<<(Ec * E2c) / 256, 256, 0, stream>>>(ffn_w2, fw216, Ec * E2c);

  auto gemm = [&](const _Float16* A, const _Float16* W, const float* bias,
                  const float* rs, float* o32, _Float16* o16, int n, int k, int gelu) {
    int strips = (M / 16) * (n / 64);  // 16x64 strip per wave, 8 waves / block
    gemm_wmma_kernel<<<strips / 8, 256, 0, stream>>>(A, W, bias, rs, o32, o16, M, n, k, gelu);
  };

  // Loop-invariant K/V projections (hoisted like the reference)
  gemm(kv16, kw16, nullptr, nullptr, nullptr, k16, Ec, Ec, 0);
  gemm(kv16, vw16, nullptr, nullptr, nullptr, s16a, Ec, Ec, 0);
  transpose_v_kernel<<<(int)(BDE / 256), 256, 0, stream>>>(s16a, vt16);

  for (int l = 0; l < Lc; ++l) {
    gemm(h16, qw16, nullptr, qsc, nullptr, s16a, Ec, Ec, 0);  // Q + FiLM scale
    flash_attn_kernel<<<(Bc * Hc * (Dc / 16)) / 8, 256, 0, stream>>>(
        s16a, k16, vt16, f32b, alibi_s);
    resid_ln_kernel<<<M, 256, 0, stream>>>(h, f32b, gamma_attn, ln1_w, ln1_b, y16);
    gemm(y16, fw116, ffn_b1, nullptr, nullptr, z16, E2c, Ec, 1);  // FFN1 + GELU
    gemm(z16, fw216, ffn_b2, nullptr, f32b, nullptr, Ec, E2c, 0); // FFN2
    resid_h16_kernel<<<(int)(BDE / 256), 256, 0, stream>>>(h, f32b, gamma_ffn, h16);
  }

  // Correction head
  resid_ln_kernel<<<M, 256, 0, stream>>>(h, nullptr, nullptr, corr_ln_w, corr_ln_b, y16);
  gemm(y16, cw116, corr_b1, nullptr, f32b, nullptr, Ec, Ec, 1);
  final_dot_kernel<<<M / 8, 256, 0, stream>>>(f32b, corr_w2, corr_b2, out);
}
